// Head_35751307772358
// MI455X (gfx1250) — compile-verified, matrix operand order checked by reading, not run on hardware
//
#include <hip/hip_runtime.h>

#define S_LEN 4096
#define BATCH 4
#define HID   512
#define HD    64
#define CHUNK 64     // keys per LDS chunk
#define PSTRIDE 72   // LDS row stride (halves) for P staging (64 + pad)

typedef __attribute__((ext_vector_type(16))) __bf16        v16bf;
typedef __attribute__((ext_vector_type(8)))  float         v8f;
typedef __attribute__((ext_vector_type(4)))  float         f32x4;
typedef __attribute__((ext_vector_type(4)))  unsigned int  u32x4;
typedef __attribute__((ext_vector_type(8)))  int           i32x8;
typedef __attribute__((ext_vector_type(4)))  int           i32x4;

union F16x16 {            // one bf16 WMMA fragment: 16 halves / lane
    v16bf   v;
    u32x4   u[2];
    __bf16  e[16];
};

__device__ __forceinline__ v8f wmma_bf16(const F16x16& a, const F16x16& b, v8f c) {
    return __builtin_amdgcn_wmma_f32_16x16x32_bf16(false, a.v, false, b.v,
                                                   (short)0, c, false, false);
}

#if __has_builtin(__builtin_amdgcn_tensor_load_to_lds) && \
    __has_builtin(__builtin_amdgcn_s_wait_tensorcnt)
#define HAVE_TDM 1
#else
#define HAVE_TDM 0
#warning "CDNA5 TDM builtins unavailable: attention uses cooperative LDS staging fallback"
#endif

#if HAVE_TDM
// Issue a 2D TDM tile load (bf16): tile (tile0 x tile1) of a tensor whose rows
// are dim0 long (stride stride0) with dim1 rows remaining from the tile start,
// from global address g to LDS byte offset lds_off.  Wave-uniform; EXEC ignored.
__device__ __forceinline__ void tdm_load_2d(unsigned lds_off, const void* g,
                                            unsigned dim0, unsigned dim1,
                                            unsigned long long stride0,
                                            unsigned tile0, unsigned tile1) {
    unsigned long long ga = (unsigned long long)(size_t)g;
    u32x4 g0;
    g0[0] = 1u;                                            // count=1, user desc
    g0[1] = lds_off;                                       // lds_addr
    g0[2] = (unsigned)(ga & 0xFFFFFFFFu);                  // global_addr[31:0]
    g0[3] = (unsigned)((ga >> 32) & 0x01FFFFFFu)           // global_addr[56:32]
          | (2u << 30);                                    // type=2 (image)
    i32x8 g1;
    g1[0] = (int)(1u << 16);                               // data_size=1 (2 B)
    g1[1] = (int)((dim0 & 0xFFFFu) << 16);                 // tensor_dim0[15:0]
    g1[2] = (int)((dim0 >> 16) | ((dim1 & 0xFFFFu) << 16));// dim0 hi | dim1 lo
    g1[3] = (int)((dim1 >> 16) | (tile0 << 16));           // dim1 hi | tile_dim0
    g1[4] = (int)(tile1 & 0xFFFFu);                        // tile_dim1 (dim2=0)
    g1[5] = (int)(stride0 & 0xFFFFFFFFu);                  // dim0_stride lo
    g1[6] = (int)((stride0 >> 32) & 0xFFFFu);              // stride hi | d1s=0
    g1[7] = 0;
    i32x4 z4 = {0, 0, 0, 0};
#if __clang_major__ >= 23
    i32x8 z8 = {0, 0, 0, 0, 0, 0, 0, 0};
    __builtin_amdgcn_tensor_load_to_lds(g0, g1, z4, z4, z8, 0);
#else
    __builtin_amdgcn_tensor_load_to_lds(g0, g1, z4, z4, 0);
#endif
}
#endif

// ---------------------------------------------------------------------------
// Kernel 0: streaming f32 -> bf16 conversion (8 elements / thread).
// ---------------------------------------------------------------------------
__global__ void cvt_f32_bf16(const float* __restrict__ src,
                             __bf16* __restrict__ dst, int n8) {
    const int i = blockIdx.x * blockDim.x + threadIdx.x;
    if (i >= n8) return;
    const float* s = src + (size_t)i * 8;
    f32x4 a = *(const f32x4*)(s);
    f32x4 b = *(const f32x4*)(s + 4);
    union { __bf16 e[8]; u32x4 u; } o;
    #pragma unroll
    for (int j = 0; j < 4; ++j) o.e[j]     = (__bf16)a[j];
    #pragma unroll
    for (int j = 0; j < 4; ++j) o.e[4 + j] = (__bf16)b[j];
    *(u32x4*)(dst + (size_t)i * 8) = o.u;
}

// ---------------------------------------------------------------------------
// Kernel 1: QKV projection, all-bf16 operands.  xb flat [16384,512] (row=s*B+b)
// times Wb[mat][64][512]^T + bias.  One wave = 16-row tile; blockIdx.y = matrix.
// Q,K -> [B][S][64];  V -> transposed [B][64][S].
// ---------------------------------------------------------------------------
__global__ void qkv_proj_wmma(const __bf16* __restrict__ xb,
                              const __bf16* __restrict__ Wb,
                              const float* __restrict__ bq,
                              const float* __restrict__ bk,
                              const float* __restrict__ bv,
                              __bf16* __restrict__ Qb,
                              __bf16* __restrict__ Kb,
                              __bf16* __restrict__ Vt) {
    const int lane = threadIdx.x & 31;
    const int wave = threadIdx.x >> 5;
    const int tile = blockIdx.x * (blockDim.x >> 5) + wave;
    const int mat  = blockIdx.y;
    const __bf16* W    = Wb + (size_t)mat * HD * HID;
    const float*  bias = (mat == 0) ? bq : (mat == 1) ? bk : bv;

    const int r0   = tile * 16;
    const int mrow = lane & 15;
    const int kb   = (lane >> 4) * 8;

    v8f acc[4];
    #pragma unroll
    for (int t = 0; t < 4; ++t)
        #pragma unroll
        for (int i = 0; i < 8; ++i) acc[t][i] = 0.0f;

    const __bf16* xr = xb + (size_t)(r0 + mrow) * HID;

    for (int ks = 0; ks < HID / 32; ++ks) {
        const int f0 = ks * 32 + kb;
        F16x16 a;
        a.u[0] = *(const u32x4*)(xr + f0);
        a.u[1] = *(const u32x4*)(xr + f0 + 16);
        #pragma unroll
        for (int t = 0; t < 4; ++t) {
            const __bf16* wr = W + (size_t)(t * 16 + mrow) * HID;
            F16x16 bfr;
            bfr.u[0] = *(const u32x4*)(wr + f0);
            bfr.u[1] = *(const u32x4*)(wr + f0 + 16);
            acc[t] = wmma_bf16(a, bfr, acc[t]);
        }
    }

    const int mbase = (lane >> 4) * 8;
    const int n     = lane & 15;
    #pragma unroll
    for (int t = 0; t < 4; ++t) {
        const int h = t * 16 + n;
        const float bval = bias[h];
        #pragma unroll
        for (int v = 0; v < 8; ++v) {
            const int r = r0 + mbase + v;
            const int b = r & (BATCH - 1);
            const int s = r >> 2;
            const __bf16 hb = (__bf16)(acc[t][v] + bval);
            if (mat == 0)      Qb[((size_t)b * S_LEN + s) * HD + h] = hb;
            else if (mat == 1) Kb[((size_t)b * S_LEN + s) * HD + h] = hb;
            else               Vt[((size_t)b * HD + h) * S_LEN + s] = hb;
        }
    }
}

// ---------------------------------------------------------------------------
// Kernel 2: flash attention.  8 waves / workgroup share one batch's K/V via
// 64-key LDS chunks, double-buffered through the Tensor Data Mover when
// available; each wave owns a 16-query tile with online softmax.
// ---------------------------------------------------------------------------
__global__ void attn_wmma(const __bf16* __restrict__ Qb,
                          const __bf16* __restrict__ Kb,
                          const __bf16* __restrict__ Vt,
                          float* __restrict__ out) {
    const int lane = threadIdx.x & 31;
    const int wave = threadIdx.x >> 5;
    const int b    = blockIdx.y;
    const int qt   = blockIdx.x * 8 + wave;
    const int q0   = qt * 16;

#if HAVE_TDM
    __shared__ __attribute__((aligned(16))) __bf16 KcS[2][CHUNK * HD]; // [key][h]
    __shared__ __attribute__((aligned(16))) __bf16 VcS[2][HD * CHUNK]; // [h][key]
#else
    __shared__ __attribute__((aligned(16))) __bf16 KcS[1][CHUNK * HD];
    __shared__ __attribute__((aligned(16))) __bf16 VcS[1][HD * CHUNK];
#endif
    __shared__ __attribute__((aligned(16))) __bf16 Pst[8][16 * PSTRIDE];
    __bf16* P = Pst[wave];

    const __bf16* Qp = Qb + (size_t)b * S_LEN * HD;
    const __bf16* Kp = Kb + (size_t)b * S_LEN * HD;
    const __bf16* Vp = Vt + (size_t)b * HD * S_LEN;

    const int mrow = lane & 15;
    const int kb   = (lane >> 4) * 8;

    // Q fragments (head dim 64 = 2 WMMA K-steps), resident all kernel
    F16x16 qa[2];
    {
        const __bf16* qr = Qp + (size_t)(q0 + mrow) * HD;
        #pragma unroll
        for (int ks = 0; ks < 2; ++ks) {
            const int h0 = ks * 32 + kb;
            qa[ks].u[0] = *(const u32x4*)(qr + h0);
            qa[ks].u[1] = *(const u32x4*)(qr + h0 + 16);
        }
    }

    v8f oacc[4];
    float rmax[8], lsum[8];
    #pragma unroll
    for (int t = 0; t < 4; ++t)
        #pragma unroll
        for (int i = 0; i < 8; ++i) oacc[t][i] = 0.0f;
    #pragma unroll
    for (int v = 0; v < 8; ++v) { rmax[v] = -3.0e38f; lsum[v] = 0.0f; }

    const float scale = 0.0441941738241592f;   // 1/sqrt(HIDDEN)
    const int nchunks = S_LEN / CHUNK;

#if HAVE_TDM
    if (wave == 0) {    // prime buffer 0 with chunk 0 via the TDM
        tdm_load_2d((unsigned)(size_t)(const void*)&KcS[0][0], Kp,
                    HD, S_LEN, HD, HD, CHUNK);
        tdm_load_2d((unsigned)(size_t)(const void*)&VcS[0][0], Vp,
                    S_LEN, HD, S_LEN, CHUNK, HD);
    }
#endif

    for (int i = 0; i < nchunks; ++i) {
        const int kv = i * CHUNK;
#if HAVE_TDM
        const int cur = i & 1;
        if (wave == 0) __builtin_amdgcn_s_wait_tensorcnt(0);
        __syncthreads();                       // chunk i resident for everyone
        if (i + 1 < nchunks && wave == 0) {    // prefetch chunk i+1 into other buf
            const int kv2 = kv + CHUNK;
            tdm_load_2d((unsigned)(size_t)(const void*)&KcS[cur ^ 1][0],
                        Kp + (size_t)kv2 * HD, HD, S_LEN - kv2, HD, HD, CHUNK);
            tdm_load_2d((unsigned)(size_t)(const void*)&VcS[cur ^ 1][0],
                        Vp + kv2, S_LEN - kv2, HD, S_LEN, CHUNK, HD);
        }
#else
        const int cur = 0;
        __syncthreads();                       // previous readers done
        for (int j = threadIdx.x; j < (CHUNK * HD) / 8; j += blockDim.x)
            *(u32x4*)&KcS[0][j * 8] = *(const u32x4*)(Kp + (size_t)kv * HD + j * 8);
        for (int j = threadIdx.x; j < (HD * CHUNK) / 8; j += blockDim.x) {
            const int h = j >> 3, c = (j & 7) * 8;
            *(u32x4*)&VcS[0][h * CHUNK + c] =
                *(const u32x4*)(Vp + (size_t)h * S_LEN + kv + c);
        }
        __syncthreads();
#endif
        const __bf16* Kc = KcS[cur];
        const __bf16* Vc = VcS[cur];

        // ---- scores: S[16q x 64k] = Q(16x64) * K^T, four 16-key sub-tiles
        v8f s[4];
        #pragma unroll
        for (int kt = 0; kt < 4; ++kt)
            #pragma unroll
            for (int j = 0; j < 8; ++j) s[kt][j] = 0.0f;
        #pragma unroll
        for (int ks = 0; ks < 2; ++ks) {
            const int h0 = ks * 32 + kb;
            #pragma unroll
            for (int kt = 0; kt < 4; ++kt) {
                const __bf16* krow = Kc + (size_t)(kt * 16 + mrow) * HD + h0;
                F16x16 bfr;
                bfr.u[0] = *(const u32x4*)(krow);
                bfr.u[1] = *(const u32x4*)(krow + 16);
                s[kt] = wmma_bf16(qa[ks], bfr, s[kt]);
            }
        }

        // ---- online softmax + O rescale + P staging (row m = v + 8*(lane/16))
        const int n = lane & 15;
        const int mq = (lane >> 4) << 3;
        #pragma unroll
        for (int v = 0; v < 8; ++v) {
            float a0 = s[0][v] * scale, a1 = s[1][v] * scale;
            float a2 = s[2][v] * scale, a3 = s[3][v] * scale;
            float mx = fmaxf(fmaxf(a0, a1), fmaxf(a2, a3));
            #pragma unroll
            for (int off = 1; off < 16; off <<= 1)
                mx = fmaxf(mx, __shfl_xor(mx, off, 32));
            const float newm = fmaxf(rmax[v], mx);
            const float c  = __expf(rmax[v] - newm);
            const float e0 = __expf(a0 - newm);
            const float e1 = __expf(a1 - newm);
            const float e2 = __expf(a2 - newm);
            const float e3 = __expf(a3 - newm);
            float rs = (e0 + e1) + (e2 + e3);
            #pragma unroll
            for (int off = 1; off < 16; off <<= 1)
                rs += __shfl_xor(rs, off, 32);
            rmax[v] = newm;
            lsum[v] = lsum[v] * c + rs;
            #pragma unroll
            for (int t = 0; t < 4; ++t) oacc[t][v] *= c;
            __bf16* pr = P + (v + mq) * PSTRIDE + n;
            pr[0]  = (__bf16)e0;
            pr[16] = (__bf16)e1;
            pr[32] = (__bf16)e2;
            pr[48] = (__bf16)e3;
        }

        // ---- O += P * V : two 32-key K-steps x four 16-col head tiles
        #pragma unroll
        for (int pks = 0; pks < 2; ++pks) {
            F16x16 pa;
            const __bf16* pr = P + mrow * PSTRIDE + pks * 32 + kb;
            pa.u[0] = *(const u32x4*)(pr);
            pa.u[1] = *(const u32x4*)(pr + 16);
            #pragma unroll
            for (int t = 0; t < 4; ++t) {
                const __bf16* vr = Vc + (size_t)(t * 16 + mrow) * CHUNK
                                      + pks * 32 + kb;
                F16x16 vb;
                vb.u[0] = *(const u32x4*)(vr);
                vb.u[1] = *(const u32x4*)(vr + 16);
                oacc[t] = wmma_bf16(pa, vb, oacc[t]);
            }
        }
#if HAVE_TDM
        __syncthreads();   // everyone done with buf[cur] before it is reissued
#endif
    }

    // ---- epilogue: divide by row sums, store [S][B][64] f32
    const int n = lane & 15;
    #pragma unroll
    for (int v = 0; v < 8; ++v) {
        const int m = v + ((lane >> 4) << 3);
        const int s = q0 + m;
        const float inv = 1.0f / lsum[v];
        #pragma unroll
        for (int t = 0; t < 4; ++t) {
            const int h = t * 16 + n;
            out[((size_t)s * BATCH + b) * HD + h] = oacc[t][v] * inv;
        }
    }
}

// ---------------------------------------------------------------------------
extern "C" void kernel_launch(void* const* d_in, const int* in_sizes, int n_in,
                              void* d_out, int out_size, void* d_ws, size_t ws_size,
                              hipStream_t stream) {
    (void)in_sizes; (void)n_in; (void)out_size; (void)ws_size;
    const float* x  = (const float*)d_in[0];
    const float* Wq = (const float*)d_in[1];
    const float* bq = (const float*)d_in[2];
    const float* Wk = (const float*)d_in[3];
    const float* bk = (const float*)d_in[4];
    const float* Wv = (const float*)d_in[5];
    const float* bv = (const float*)d_in[6];
    float* out = (float*)d_out;

    const size_t QKV = (size_t)BATCH * S_LEN * HD;        // 1 Mi elements
    __bf16* Qb = (__bf16*)d_ws;                           // [B][S][64]
    __bf16* Kb = Qb + QKV;                                // [B][S][64]
    __bf16* Vt = Kb + QKV;                                // [B][64][S]
    __bf16* xb = Vt + QKV;                                // [16384][512]
    __bf16* Wb = xb + (size_t)S_LEN * BATCH * HID;        // [3][64][512]

    // f32 -> bf16 streaming converts (x, then the three weight matrices)
    {
        const int n8x = (S_LEN * BATCH * HID) / 8;
        cvt_f32_bf16<<<(n8x + 255) / 256, 256, 0, stream>>>(x, xb, n8x);
        const int n8w = (HD * HID) / 8;
        const int blk = (n8w + 255) / 256;
        cvt_f32_bf16<<<blk, 256, 0, stream>>>(Wq, Wb,               n8w);
        cvt_f32_bf16<<<blk, 256, 0, stream>>>(Wk, Wb + HD * HID,    n8w);
        cvt_f32_bf16<<<blk, 256, 0, stream>>>(Wv, Wb + 2 * HD * HID, n8w);
    }

    dim3 gproj(256, 3);      // 16384 rows / 16 per wave / 4 waves per block
    qkv_proj_wmma<<<gproj, 128, 0, stream>>>(xb, Wb, bq, bk, bv, Qb, Kb, Vt);

    dim3 gattn(32, BATCH);   // 256 q-tiles per batch / 8 waves per block
    attn_wmma<<<gattn, 256, 0, stream>>>(Qb, Kb, Vt, out);
}
